// KPConv_29970281791917
// MI455X (gfx1250) — matrix operand order repaired and checked by reading, not compiled, not run on hardware
//
#include <hip/hip_runtime.h>
#include <hip/hip_bf16.h>

#define B_    4
#define N_    4096
#define CIN   64
#define COUT  128
#define KP    15
#define KNN_  16
#define KTOT  (KP * CIN)      // 960 = reduction dim of the big GEMM
#define SIG2INV 100.0f        // 1 / (0.1*0.1)

typedef __attribute__((ext_vector_type(16))) _Float16 v16h;
typedef __attribute__((ext_vector_type(8)))  float    v8f;

union FragU { v16h v; uint4 q[2]; };

// ---------------------------------------------------------------------------
// Kernel 1: brute-force KNN (exclude self), 1 thread per query point.
// Candidate coords streamed through LDS in tiles of 128; top-16 kept sorted
// ascending in registers via a fully unrolled insertion network.
// ---------------------------------------------------------------------------
__global__ __launch_bounds__(128)
void knn_kernel(const float* __restrict__ coords, int* __restrict__ nbr) {
  const int b = blockIdx.y;
  const int i = blockIdx.x * 128 + threadIdx.x;
  const float* cb = coords + (size_t)b * 3 * N_;
  const float xi = cb[i], yi = cb[N_ + i], zi = cb[2 * N_ + i];

  float bd[KNN_]; int bi[KNN_];
#pragma unroll
  for (int s = 0; s < KNN_; ++s) { bd[s] = 3.0e38f; bi[s] = 0; }

  __shared__ float tx[128], ty[128], tz[128];
  for (int j0 = 0; j0 < N_; j0 += 128) {
    __syncthreads();
    const int t = threadIdx.x;
    tx[t] = cb[j0 + t];
    ty[t] = cb[N_ + j0 + t];
    tz[t] = cb[2 * N_ + j0 + t];
    __syncthreads();
    for (int jj = 0; jj < 128; ++jj) {
      const int j = j0 + jj;
      const float dx = xi - tx[jj], dy = yi - ty[jj], dz = zi - tz[jj];
      float d = dx * dx + dy * dy + dz * dz;
      if (j == i) d = 3.0e38f;          // exclude self
      if (d < bd[KNN_ - 1]) {
#pragma unroll
        for (int s = KNN_ - 1; s >= 0; --s) {   // descending: reads of bd[s-1]
          if (d < bd[s]) {                      // happen before writes to it
            const float pd = (s == 0) ? -1.0f : bd[s - 1];
            const int   pi = (s == 0) ? 0     : bi[s - 1];
            const bool shift = d < pd;          // d>=0 so never shifts at s==0
            bd[s] = shift ? pd : d;
            bi[s] = shift ? pi : j;
          }
        }
      }
    }
  }
  int* onb = nbr + ((size_t)b * N_ + i) * KNN_;
#pragma unroll
  for (int s = 0; s < KNN_; ++s) onb[s] = bi[s];
}

// ---------------------------------------------------------------------------
// Kernel 2: weights (p,c,o) fp32 -> f16, transposed to (o, k=p*64+c) so each
// lane's WMMA B-fragment is two contiguous 16B loads.
// ---------------------------------------------------------------------------
__global__ void prep_weights(const float* __restrict__ kw, _Float16* __restrict__ wt) {
  const int t = blockIdx.x * 256 + threadIdx.x;
  if (t >= KTOT * COUT) return;
  const int k = t % KTOT;        // p*64 + c
  const int o = t / KTOT;
  wt[(size_t)o * KTOT + k] = (_Float16)kw[(size_t)k * COUT + o];
}

// ---------------------------------------------------------------------------
// Kernel 3: fused influence + aggregation + output GEMM.
// 16 points per block (one WMMA M-tile), 256 threads = 8 waves, each wave
// owns 16 output channels (COUT=128). K = 960 -> 30 chunks of
// v_wmma_f32_16x16x32_f16, fp32 accumulation.
// ---------------------------------------------------------------------------
__global__ __launch_bounds__(256)
void kpconv_kernel(const float* __restrict__ coords,
                   const float* __restrict__ feats,
                   const float* __restrict__ kpts,
                   const float* __restrict__ bias,
                   const int*   __restrict__ nbr,
                   const _Float16* __restrict__ wt,
                   float* __restrict__ out) {
  // [0,32768):  phase1 nf f16[16][16][64]  -> phase3 agg f16[16][960] (aliased)
  // [32768,48128): infl f32[16][16][15]
  __shared__ __align__(16) unsigned char smem[48128];
  _Float16* nf   = (_Float16*)smem;
  float*    infl = (float*)(smem + 32768);

  const int b   = blockIdx.y;
  const int n0  = blockIdx.x * 16;
  const int tid = threadIdx.x;

  // ---- Phase 1: influences + f16 feature gather (1 thread per (m,k)) ----
  {
    const int m = tid >> 4, k = tid & 15;
    const int i = n0 + m;
    const float* cb  = coords + (size_t)b * 3 * N_;
    const int    idx = nbr[((size_t)b * N_ + i) * KNN_ + k];
    const float dx = cb[idx]          - cb[i];
    const float dy = cb[N_ + idx]     - cb[N_ + i];
    const float dz = cb[2 * N_ + idx] - cb[2 * N_ + i];
#pragma unroll
    for (int p = 0; p < KP; ++p) {
      const float ex = dx - kpts[p * 3 + 0];
      const float ey = dy - kpts[p * 3 + 1];
      const float ez = dz - kpts[p * 3 + 2];
      const float sq = ex * ex + ey * ey + ez * ez;
      infl[(m * 16 + k) * KP + p] = __expf(-sq * SIG2INV);
    }
    const float* fb = feats + (size_t)b * CIN * N_ + idx;
    for (int c = 0; c < CIN; ++c)
      nf[(m * 16 + k) * CIN + c] = (_Float16)fb[(size_t)c * N_];
  }
  __syncthreads();

  // ---- Phase 2: agg[m][p*64+c] = sum_k infl[m][k][p] * nf[m][k][c] ----
  // Thread owns channel c for 4 points; results held in regs until all
  // nf reads are done (agg aliases nf).
  _Float16 rres[4][KP];
  {
    const int c = tid & 63, g = tid >> 6;
#pragma unroll
    for (int mm = 0; mm < 4; ++mm) {
      const int m = g * 4 + mm;
      float s[KP];
#pragma unroll
      for (int p = 0; p < KP; ++p) s[p] = 0.0f;
      for (int k = 0; k < KNN_; ++k) {
        const float f = (float)nf[(m * 16 + k) * CIN + c];
#pragma unroll
        for (int p = 0; p < KP; ++p) s[p] += infl[(m * 16 + k) * KP + p] * f;
      }
#pragma unroll
      for (int p = 0; p < KP; ++p) rres[mm][p] = (_Float16)s[p];
    }
  }
  __syncthreads();
  {
    _Float16* agg = (_Float16*)smem;
    const int c = tid & 63, g = tid >> 6;
#pragma unroll
    for (int mm = 0; mm < 4; ++mm) {
      const int m = g * 4 + mm;
#pragma unroll
      for (int p = 0; p < KP; ++p) agg[m * KTOT + p * CIN + c] = rres[mm][p];
    }
  }
  __syncthreads();

  // ---- Phase 3: WMMA GEMM: D[16 x 16] += A[16 x 960] * B[960 x 16] ----
  {
    const _Float16* agg = (const _Float16*)smem;
    const int lane = tid & 31;
    const int w    = tid >> 5;               // wave id -> 16 output channels
    const int half = (lane >> 4) & 1;        // lane group (0-15 / 16-31)
    const int mA   = lane & 15;              // A row (point in tile)
    const int o    = w * 16 + (lane & 15);   // output channel
    const _Float16* wrow = wt + (size_t)o * KTOT;

    v8f acc = {};
    for (int kc = 0; kc < KTOT / 32; ++kc) {
      FragU a, bfr;
      // A (16-bit 16x32 layout): halves 0-7 = K0..K0+7, halves 8-15 = K0+16..K0+23
      const int K0 = kc * 32 + half * 8;
      a.q[0] = *(const uint4*)(agg + mA * KTOT + K0);
      a.q[1] = *(const uint4*)(agg + mA * KTOT + K0 + 16);
      // B (32x16): lane = column, same K order as A: halves 0-7 = K0..K0+7, halves 8-15 = K0+16..K0+23
      const int KB = kc * 32 + half * 8;
      bfr.q[0] = *(const uint4*)(wrow + KB);
      bfr.q[1] = *(const uint4*)(wrow + KB + 16);
      acc = __builtin_amdgcn_wmma_f32_16x16x32_f16(
          /*neg_a=*/false, a.v, /*neg_b=*/false, bfr.v,
          /*c_mod=*/(short)0, acc, /*reuse_a=*/false, /*reuse_b=*/false);
    }

    // C/D layout: VGPR r -> M = r + half*8, N(col=o) = lane&15.
    const float bo = bias[o];
    float* orow = out + ((size_t)b * COUT + o) * N_ + n0 + half * 8;
#pragma unroll
    for (int r = 0; r < 8; ++r) orow[r] = acc[r] + bo;
  }
}

// ---------------------------------------------------------------------------
extern "C" void kernel_launch(void* const* d_in, const int* in_sizes, int n_in,
                              void* d_out, int out_size, void* d_ws, size_t ws_size,
                              hipStream_t stream) {
  const float* coords = (const float*)d_in[0];   // (B,3,N)
  const float* feats  = (const float*)d_in[1];   // (B,C_in,N)
  const float* kpts   = (const float*)d_in[2];   // (15,3)
  const float* kw     = (const float*)d_in[3];   // (15,64,128)
  const float* bias   = (const float*)d_in[4];   // (128,)

  int*      nbr = (int*)d_ws;                                            // 1 MB
  _Float16* wt  = (_Float16*)((char*)d_ws +
                              (size_t)B_ * N_ * KNN_ * sizeof(int));     // 240 KB

  knn_kernel<<<dim3(N_ / 128, B_), 128, 0, stream>>>(coords, nbr);
  prep_weights<<<(KTOT * COUT + 255) / 256, 256, 0, stream>>>(kw, wt);
  kpconv_kernel<<<dim3(N_ / 16, B_), 256, 0, stream>>>(
      coords, feats, kpts, bias, nbr, wt, (float*)d_out);
}
